// E_GCL_44109314129961
// MI455X (gfx1250) — compile-verified, matrix operand order checked by reading, not run on hardware
//
#include <hip/hip_runtime.h>
#include <cstdint>

// ---------------------------------------------------------------------------
// Types / helpers for CDNA5 WMMA (wave32, 16x16x32 bf16 -> f32)
// ---------------------------------------------------------------------------
typedef __attribute__((ext_vector_type(16))) __bf16 v16bf;
typedef __attribute__((ext_vector_type(8)))  float  v8f;

union BFrag {
    unsigned short u[16];
    uint4          q[2];
    v16bf          b;
};

// Hardware f32 -> bf16 convert (v_cvt on gfx1250), instead of software RNE.
__device__ __forceinline__ unsigned short f2bf(float x) {
    union { __bf16 h; unsigned short u; } c;
    c.h = (__bf16)x;
    return c.u;
}

__device__ __forceinline__ float silu_f(float x) {
    return x / (1.0f + __expf(-x));
}

__device__ __forceinline__ v8f zero8() {
    v8f z;
#pragma unroll
    for (int i = 0; i < 8; ++i) z[i] = 0.0f;
    return z;
}

__device__ __forceinline__ v8f wmma_bf16(v16bf a, v16bf b, v8f c) {
    // 8 args: (neg_a, A, neg_b, B, c_mod, C, reuse_a, reuse_b)
    return __builtin_amdgcn_wmma_f32_16x16x32_bf16(false, a, false, b, (short)0, c,
                                                   false, false);
}

// A fragment (16x32 bf16) gathered from a global fp32 row.
// Lane layout (ISA 16-bit A 16x32): lane holds row m = lane&15;
// khi = lane>>4; element i -> k = k0 + (i&7) + (i>>3)*16, k0 = ks*32 + khi*8.
__device__ __forceinline__ v16bf gather_afrag_f32(const float* src, int k0) {
    float t[16];
    float4 a0 = *(const float4*)(src + k0);
    float4 a1 = *(const float4*)(src + k0 + 4);
    float4 b0 = *(const float4*)(src + k0 + 16);
    float4 b1 = *(const float4*)(src + k0 + 20);
    t[0]=a0.x;  t[1]=a0.y;  t[2]=a0.z;  t[3]=a0.w;
    t[4]=a1.x;  t[5]=a1.y;  t[6]=a1.z;  t[7]=a1.w;
    t[8]=b0.x;  t[9]=b0.y;  t[10]=b0.z; t[11]=b0.w;
    t[12]=b1.x; t[13]=b1.y; t[14]=b1.z; t[15]=b1.w;
    BFrag f;
#pragma unroll
    for (int i = 0; i < 16; ++i) f.u[i] = f2bf(t[i]);
    return f.b;
}

// A fragment read from bf16 staged in LDS (row = &stage[m*128]).
__device__ __forceinline__ v16bf load_afrag_lds(const unsigned short* row, int k0) {
    BFrag f;
    f.q[0] = *(const uint4*)(row + k0);
    f.q[1] = *(const uint4*)(row + k0 + 16);
    return f.b;
}

// B fragment from packed weights: each 512-element block holds one (ntile,kstep)
// fragment, 16 contiguous bf16 per lane.
__device__ __forceinline__ v16bf load_bfrag(const unsigned short* blk, int lane) {
    BFrag f;
    const uint4* q = (const uint4*)(blk + lane * 16);
    f.q[0] = q[0];
    f.q[1] = q[1];
    return f.b;
}

// ---------------------------------------------------------------------------
// Kernel 0: zero the accumulators (h_agg, agg_sum, cnt are contiguous)
// ---------------------------------------------------------------------------
__global__ void zero_kernel(float* p, int n) {
    int i = blockIdx.x * blockDim.x + threadIdx.x;
    if (i < n) p[i] = 0.0f;
}

// ---------------------------------------------------------------------------
// Kernel 1: convert fp32 weights to bf16 in B-fragment-packed layout.
// 7 logical 128x128 matrices: We1[0:128], We1[128:256], We2, Wc1,
//                             Wn1[0:128], Wn1[128:256], Wn2.
// Block order within a matrix: blk = ntile*4 + kstep (8 ntiles x 4 ksteps).
// Within a block: [lane][i] with k = kt*32 + (lane>>4)*8 + (i&7) + (i>>3)*16,
//                                n = nt*16 + (lane&15).
// ---------------------------------------------------------------------------
__global__ void pack_weights(const float* We1, const float* We2, const float* Wc1,
                             const float* Wn1, const float* Wn2,
                             unsigned short* packed) {
    const int PER = 8 * 4 * 512;   // 16384 per matrix
    int tid = blockIdx.x * blockDim.x + threadIdx.x;
    if (tid >= 7 * PER) return;
    int mat = tid / PER;
    int e   = tid % PER;
    int blk = e >> 9;
    int l   = (e >> 4) & 31;
    int i   = e & 15;
    int nt = blk >> 2, kt = blk & 3;
    int lo = l & 15, khi = l >> 4;
    int k = kt * 32 + khi * 8 + (i & 7) + ((i >> 3) << 4);
    int n = nt * 16 + lo;
    float v;
    switch (mat) {
        case 0:  v = We1[k * 128 + n];           break;
        case 1:  v = We1[(k + 128) * 128 + n];   break;
        case 2:  v = We2[k * 128 + n];           break;
        case 3:  v = Wc1[k * 128 + n];           break;
        case 4:  v = Wn1[k * 128 + n];           break;
        case 5:  v = Wn1[(k + 128) * 128 + n];   break;
        default: v = Wn2[k * 128 + n];           break;
    }
    packed[tid] = f2bf(v);
}

// ---------------------------------------------------------------------------
// Kernel 2: edge pipeline. One wave per 32-edge tile (two 16x16 M-tiles),
// so each B fragment is reused by two WMMAs; B fragments prefetched in
// batches of 4 so loads clause together ahead of the WMMA burst.
// ---------------------------------------------------------------------------
__global__ __launch_bounds__(32)
void egcl_edge_kernel(const float* __restrict__ h, const float* __restrict__ y,
                      const int* __restrict__ ei,
                      const float* __restrict__ We1,   // for radial row 256
                      const float* __restrict__ be1, const float* __restrict__ be2,
                      const float* __restrict__ bc1, const float* __restrict__ Wc2,
                      const unsigned short* __restrict__ Pe1a,
                      const unsigned short* __restrict__ Pe1b,
                      const unsigned short* __restrict__ Pe2,
                      const unsigned short* __restrict__ Pc1,
                      float* __restrict__ h_agg, float* __restrict__ agg_sum,
                      float* __restrict__ cnt, int E) {
    __shared__ int            s_row[32];
    __shared__ int            s_col[32];
    __shared__ float          s_diff[32][3];
    __shared__ float          s_radial[32];
    __shared__ float          s_sred[32];
    __shared__ unsigned short s_stage[32 * 128];   // 8 KB bf16 activation tile

    const int lane  = threadIdx.x;
    const int ebase = blockIdx.x * 32;
    const int lo = lane & 15;
    const int hi = lane >> 4;
    const int mb = hi * 8;          // C-fragment row base within a 16-row tile

    {
        int e = ebase + lane;
        int r = ei[e];
        int c = ei[E + e];
        s_row[lane] = r;
        s_col[lane] = c;
        float dx = y[r * 3 + 0] - y[c * 3 + 0];
        float dy = y[r * 3 + 1] - y[c * 3 + 1];
        float dz = y[r * 3 + 2] - y[c * 3 + 2];
        s_diff[lane][0] = dx; s_diff[lane][1] = dy; s_diff[lane][2] = dz;
        s_radial[lane] = dx * dx + dy * dy + dz * dz;
        s_sred[lane] = 0.0f;
    }
    __syncthreads();

    const float* h0r = h + (long)s_row[lo] * 128;        // tile 0 rows
    const float* h1r = h + (long)s_row[16 + lo] * 128;   // tile 1 rows
    const float* h0c = h + (long)s_col[lo] * 128;
    const float* h1c = h + (long)s_col[16 + lo] * 128;

    v8f C0[8], C1[8];
#pragma unroll
    for (int n = 0; n < 8; ++n) { C0[n] = zero8(); C1[n] = zero8(); }

    // ---- layer 1: pre1 = h_row@We1a + h_col@We1b + radial*We1[256] + be1 ----
#pragma unroll
    for (int ks = 0; ks < 4; ++ks) {
        v16bf a0 = gather_afrag_f32(h0r, ks * 32 + hi * 8);
        v16bf a1 = gather_afrag_f32(h1r, ks * 32 + hi * 8);
#pragma unroll
        for (int nh = 0; nh < 2; ++nh) {
            v16bf B[4];
#pragma unroll
            for (int j = 0; j < 4; ++j)
                B[j] = load_bfrag(Pe1a + ((nh * 4 + j) * 4 + ks) * 512, lane);
#pragma unroll
            for (int j = 0; j < 4; ++j) {
                C0[nh * 4 + j] = wmma_bf16(a0, B[j], C0[nh * 4 + j]);
                C1[nh * 4 + j] = wmma_bf16(a1, B[j], C1[nh * 4 + j]);
            }
        }
    }
#pragma unroll
    for (int ks = 0; ks < 4; ++ks) {
        v16bf a0 = gather_afrag_f32(h0c, ks * 32 + hi * 8);
        v16bf a1 = gather_afrag_f32(h1c, ks * 32 + hi * 8);
#pragma unroll
        for (int nh = 0; nh < 2; ++nh) {
            v16bf B[4];
#pragma unroll
            for (int j = 0; j < 4; ++j)
                B[j] = load_bfrag(Pe1b + ((nh * 4 + j) * 4 + ks) * 512, lane);
#pragma unroll
            for (int j = 0; j < 4; ++j) {
                C0[nh * 4 + j] = wmma_bf16(a0, B[j], C0[nh * 4 + j]);
                C1[nh * 4 + j] = wmma_bf16(a1, B[j], C1[nh * 4 + j]);
            }
        }
    }

    float rad0[8], rad1[8];
#pragma unroll
    for (int r = 0; r < 8; ++r) {
        rad0[r] = s_radial[mb + r];
        rad1[r] = s_radial[16 + mb + r];
    }

#pragma unroll
    for (int n = 0; n < 8; ++n) {
        int   colx = n * 16 + lo;
        float b1   = be1[colx];
        float w256 = We1[256 * 128 + colx];
#pragma unroll
        for (int r = 0; r < 8; ++r) {
            float v0 = silu_f(C0[n][r] + b1 + rad0[r] * w256);
            float v1 = silu_f(C1[n][r] + b1 + rad1[r] * w256);
            s_stage[(mb + r) * 128 + colx]        = f2bf(v0);
            s_stage[(16 + mb + r) * 128 + colx]   = f2bf(v1);
        }
    }
    __syncthreads();

    // ---- layer 2: edge_feat = silu(x1 @ We2 + be2) ----
#pragma unroll
    for (int n = 0; n < 8; ++n) { C0[n] = zero8(); C1[n] = zero8(); }
#pragma unroll
    for (int ks = 0; ks < 4; ++ks) {
        v16bf a0 = load_afrag_lds(&s_stage[lo * 128],        ks * 32 + hi * 8);
        v16bf a1 = load_afrag_lds(&s_stage[(16 + lo) * 128], ks * 32 + hi * 8);
#pragma unroll
        for (int nh = 0; nh < 2; ++nh) {
            v16bf B[4];
#pragma unroll
            for (int j = 0; j < 4; ++j)
                B[j] = load_bfrag(Pe2 + ((nh * 4 + j) * 4 + ks) * 512, lane);
#pragma unroll
            for (int j = 0; j < 4; ++j) {
                C0[nh * 4 + j] = wmma_bf16(a0, B[j], C0[nh * 4 + j]);
                C1[nh * 4 + j] = wmma_bf16(a1, B[j], C1[nh * 4 + j]);
            }
        }
    }
    __syncthreads();   // stage reads done before overwrite

    int rn0[8], rn1[8];
#pragma unroll
    for (int r = 0; r < 8; ++r) {
        rn0[r] = s_row[mb + r];
        rn1[r] = s_row[16 + mb + r];
    }

#pragma unroll
    for (int n = 0; n < 8; ++n) {
        int   colx = n * 16 + lo;
        float b2   = be2[colx];
#pragma unroll
        for (int r = 0; r < 8; ++r) {
            float v0 = silu_f(C0[n][r] + b2);
            float v1 = silu_f(C1[n][r] + b2);
            s_stage[(mb + r) * 128 + colx]      = f2bf(v0);
            s_stage[(16 + mb + r) * 128 + colx] = f2bf(v1);
            atomicAdd(&h_agg[(long)rn0[r] * 128 + colx], v0);   // segment_sum
            atomicAdd(&h_agg[(long)rn1[r] * 128 + colx], v1);
        }
    }
    __syncthreads();

    // ---- coord MLP: s = silu(edge_feat @ Wc1 + bc1) @ Wc2 ----
#pragma unroll
    for (int n = 0; n < 8; ++n) { C0[n] = zero8(); C1[n] = zero8(); }
#pragma unroll
    for (int ks = 0; ks < 4; ++ks) {
        v16bf a0 = load_afrag_lds(&s_stage[lo * 128],        ks * 32 + hi * 8);
        v16bf a1 = load_afrag_lds(&s_stage[(16 + lo) * 128], ks * 32 + hi * 8);
#pragma unroll
        for (int nh = 0; nh < 2; ++nh) {
            v16bf B[4];
#pragma unroll
            for (int j = 0; j < 4; ++j)
                B[j] = load_bfrag(Pc1 + ((nh * 4 + j) * 4 + ks) * 512, lane);
#pragma unroll
            for (int j = 0; j < 4; ++j) {
                C0[nh * 4 + j] = wmma_bf16(a0, B[j], C0[nh * 4 + j]);
                C1[nh * 4 + j] = wmma_bf16(a1, B[j], C1[nh * 4 + j]);
            }
        }
    }

    float p0[8], p1[8];
#pragma unroll
    for (int r = 0; r < 8; ++r) { p0[r] = 0.0f; p1[r] = 0.0f; }
#pragma unroll
    for (int n = 0; n < 8; ++n) {
        int   colx = n * 16 + lo;
        float b3   = bc1[colx];
        float w2   = Wc2[colx];
#pragma unroll
        for (int r = 0; r < 8; ++r) {
            p0[r] += silu_f(C0[n][r] + b3) * w2;
            p1[r] += silu_f(C1[n][r] + b3) * w2;
        }
    }
#pragma unroll
    for (int r = 0; r < 8; ++r) {
        atomicAdd(&s_sred[mb + r],      p0[r]);
        atomicAdd(&s_sred[16 + mb + r], p1[r]);
    }
    __syncthreads();

    {
        float s    = s_sred[lane];
        int   ridx = s_row[lane];
        atomicAdd(&agg_sum[ridx * 3 + 0], s_diff[lane][0] * s);
        atomicAdd(&agg_sum[ridx * 3 + 1], s_diff[lane][1] * s);
        atomicAdd(&agg_sum[ridx * 3 + 2], s_diff[lane][2] * s);
        atomicAdd(&cnt[ridx], 1.0f);
    }
}

// ---------------------------------------------------------------------------
// Kernel 3: node pipeline + coordinate update. One wave per 16-node tile.
// ---------------------------------------------------------------------------
__global__ __launch_bounds__(32)
void egcl_node_kernel(const float* __restrict__ h, const float* __restrict__ y,
                      const float* __restrict__ bn1, const float* __restrict__ bn2,
                      const unsigned short* __restrict__ Pn1a,
                      const unsigned short* __restrict__ Pn1b,
                      const unsigned short* __restrict__ Pn2,
                      const float* __restrict__ h_agg,
                      const float* __restrict__ agg_sum,
                      const float* __restrict__ cnt,
                      float* __restrict__ h_out, float* __restrict__ y_out) {
    __shared__ unsigned short s_stage[16 * 128];

    const int lane  = threadIdx.x;
    const int nbase = blockIdx.x * 16;
    const int lo = lane & 15;
    const int hi = lane >> 4;
    const int mb = hi * 8;

    const float* hrow = h     + (long)(nbase + lo) * 128;
    const float* arow = h_agg + (long)(nbase + lo) * 128;

    v8f C[8];
#pragma unroll
    for (int n = 0; n < 8; ++n) C[n] = zero8();

#pragma unroll
    for (int ks = 0; ks < 4; ++ks) {
        v16bf a = gather_afrag_f32(hrow, ks * 32 + hi * 8);
        v16bf B[8];
#pragma unroll
        for (int n = 0; n < 8; ++n)
            B[n] = load_bfrag(Pn1a + (n * 4 + ks) * 512, lane);
#pragma unroll
        for (int n = 0; n < 8; ++n)
            C[n] = wmma_bf16(a, B[n], C[n]);
    }
#pragma unroll
    for (int ks = 0; ks < 4; ++ks) {
        v16bf a = gather_afrag_f32(arow, ks * 32 + hi * 8);
        v16bf B[8];
#pragma unroll
        for (int n = 0; n < 8; ++n)
            B[n] = load_bfrag(Pn1b + (n * 4 + ks) * 512, lane);
#pragma unroll
        for (int n = 0; n < 8; ++n)
            C[n] = wmma_bf16(a, B[n], C[n]);
    }

#pragma unroll
    for (int n = 0; n < 8; ++n) {
        int   colx = n * 16 + lo;
        float b1   = bn1[colx];
#pragma unroll
        for (int r = 0; r < 8; ++r)
            s_stage[(mb + r) * 128 + colx] = f2bf(silu_f(C[n][r] + b1));
    }
    __syncthreads();

#pragma unroll
    for (int n = 0; n < 8; ++n) C[n] = zero8();
#pragma unroll
    for (int ks = 0; ks < 4; ++ks) {
        v16bf a = load_afrag_lds(&s_stage[lo * 128], ks * 32 + hi * 8);
        v16bf B[8];
#pragma unroll
        for (int n = 0; n < 8; ++n)
            B[n] = load_bfrag(Pn2 + (n * 4 + ks) * 512, lane);
#pragma unroll
        for (int n = 0; n < 8; ++n)
            C[n] = wmma_bf16(a, B[n], C[n]);
    }

#pragma unroll
    for (int n = 0; n < 8; ++n) {
        int   colx = n * 16 + lo;
        float b2   = bn2[colx];
#pragma unroll
        for (int r = 0; r < 8; ++r) {
            long idx = (long)(nbase + mb + r) * 128 + colx;
            h_out[idx] = h[idx] + C[n][r] + b2;    // residual
        }
    }

    if (lane < 16) {
        int   node = nbase + lane;
        float c    = cnt[node];
        c = c < 1.0f ? 1.0f : c;
#pragma unroll
        for (int d = 0; d < 3; ++d)
            y_out[node * 3 + d] = y[node * 3 + d] + agg_sum[node * 3 + d] / c;
    }
}

// ---------------------------------------------------------------------------
// Host-side launcher
// ---------------------------------------------------------------------------
extern "C" void kernel_launch(void* const* d_in, const int* in_sizes, int n_in,
                              void* d_out, int out_size, void* d_ws, size_t ws_size,
                              hipStream_t stream) {
    const float* h    = (const float*)d_in[0];
    const float* y    = (const float*)d_in[1];
    const int*   ei   = (const int*)d_in[2];
    const float* We1  = (const float*)d_in[3];
    const float* be1  = (const float*)d_in[4];
    const float* We2  = (const float*)d_in[5];
    const float* be2  = (const float*)d_in[6];
    const float* Wc1  = (const float*)d_in[7];
    const float* bc1  = (const float*)d_in[8];
    const float* Wc2  = (const float*)d_in[9];
    const float* Wn1  = (const float*)d_in[10];
    const float* bn1  = (const float*)d_in[11];
    const float* Wn2  = (const float*)d_in[12];
    const float* bn2  = (const float*)d_in[13];

    const int N = in_sizes[0] / 128;
    const int E = in_sizes[2] / 2;

    // workspace layout
    float*          h_agg   = (float*)d_ws;              // N*128
    float*          agg_sum = h_agg + (long)N * 128;     // N*3
    float*          cntp    = agg_sum + (long)N * 3;     // N
    unsigned short* packed  = (unsigned short*)(cntp + N);
    const int PER = 16384;
    const unsigned short* Pe1a = packed + 0 * PER;
    const unsigned short* Pe1b = packed + 1 * PER;
    const unsigned short* Pe2  = packed + 2 * PER;
    const unsigned short* Pc1  = packed + 3 * PER;
    const unsigned short* Pn1a = packed + 4 * PER;
    const unsigned short* Pn1b = packed + 5 * PER;
    const unsigned short* Pn2  = packed + 6 * PER;

    float* h_out = (float*)d_out;
    float* y_out = h_out + (long)N * 128;

    int zn = N * 132;   // h_agg + agg_sum + cnt (contiguous)
    zero_kernel<<<(zn + 255) / 256, 256, 0, stream>>>(h_agg, zn);
    pack_weights<<<(7 * PER + 255) / 256, 256, 0, stream>>>(We1, We2, Wc1, Wn1, Wn2,
                                                            packed);
    egcl_edge_kernel<<<E / 32, 32, 0, stream>>>(h, y, ei, We1, be1, be2, bc1, Wc2,
                                                Pe1a, Pe1b, Pe2, Pc1,
                                                h_agg, agg_sum, cntp, E);
    egcl_node_kernel<<<N / 16, 32, 0, stream>>>(h, y, bn1, bn2, Pn1a, Pn1b, Pn2,
                                                h_agg, agg_sum, cntp, h_out, y_out);
}
